// TransEncoder_10058813407632
// MI455X (gfx1250) — matrix operand order repaired and checked by reading, not compiled, not run on hardware
//
#include <hip/hip_runtime.h>
#include <cstddef>
#include <cstdint>

// ---------------------------------------------------------------------------
// Transformer encoder (B=8,S=2048,D=256,H=8,DK=32,DFF=1024,L=5) for gfx1250.
// bf16 WMMA for all GEMMs (32x32 per-wave tiles), flash-attention with online
// softmax, double-buffered async global->LDS staging (ASYNCcnt, one barrier
// per K-step), fp32 residual/LN.
// Layout: weights pre-transposed [N][K]; V produced transposed [B,H,DK,S] so
// every WMMA B-fragment is a contiguous 32B LDS load.
// ---------------------------------------------------------------------------

typedef __bf16 bf16;
typedef __attribute__((ext_vector_type(16))) __bf16 v16bf;
typedef __attribute__((ext_vector_type(8)))  float  v8f;

#define TE_B   8
#define TE_S   2048
#define TE_D   256
#define TE_H   8
#define TE_DK  32
#define TE_DFF 1024
#define TE_L   5
#define TE_M   (TE_B * TE_S)          // 16384 token rows
#define TE_EPS 1e-5f

// gfx1250 async global->LDS copy (16B per lane), tracked by ASYNCcnt.
__device__ __forceinline__ void te_async_b128(const void* lds, const void* gp) {
  asm volatile("global_load_async_to_lds_b128 %0, %1, off"
               :: "v"((unsigned)(size_t)lds),
                  "v"((unsigned long long)(size_t)gp)
               : "memory");
}
__device__ __forceinline__ void te_wait_async() {
  asm volatile("s_wait_asynccnt 0" ::: "memory");
}

// ---------------------------------------------------------------------------
// fp32 -> bf16 conversion (grid-stride)
// ---------------------------------------------------------------------------
__global__ __launch_bounds__(256) void te_cvt(const float* __restrict__ in,
                                              bf16* __restrict__ out, int n) {
  for (int i = blockIdx.x * blockDim.x + threadIdx.x; i < n;
       i += gridDim.x * blockDim.x) {
    out[i] = (bf16)in[i];
  }
}

// fp32 [L][K][N] -> bf16 transposed [L][N][K]
__global__ __launch_bounds__(256) void te_cvt_t(const float* __restrict__ in,
                                                bf16* __restrict__ out,
                                                int K, int N) {
  const int total = TE_L * K * N;
  for (int i = blockIdx.x * blockDim.x + threadIdx.x; i < total;
       i += gridDim.x * blockDim.x) {
    const int l = i / (K * N);
    const int r = i - l * (K * N);
    const int n = r / K;
    const int k = r - n * K;
    out[i] = (bf16)in[(size_t)l * K * N + (size_t)k * N + n];
  }
}

// ---------------------------------------------------------------------------
// Tiled WMMA GEMM:  C[M,N] = act(A[M,K] @ W[K,N] + bias[N])
//   A bf16 row-major [M][K]; Wt bf16 pre-transposed [N][K].
//   Cf (fp32) and/or Cb (bf16).  act: 0=id 1=exact-gelu.
//   cbmode: 0 = Cb row-major [M][N]; 1 = Cb as V-transposed [B,H,DK,S].
//   WG: 256 threads / 8 waves. 128x64 WG tile, 32x32 per wave (4 WMMA/K-step),
//   K stepped by 32, double-buffered async global->LDS staging.
// ---------------------------------------------------------------------------
__global__ __launch_bounds__(256) void te_gemm(const bf16* __restrict__ A,
                                               const bf16* __restrict__ Wt,
                                               const float* __restrict__ bias,
                                               float* __restrict__ Cf,
                                               bf16* __restrict__ Cb,
                                               int M, int N, int K,
                                               int act, int cbmode) {
  __shared__ __align__(32) bf16 As[2][128 * 32];   // [m][k]  2 x 8KB
  __shared__ __align__(32) bf16 Bst[2][64 * 32];   // [n][k]  2 x 4KB

  const int tid  = threadIdx.x;
  const int lane = tid & 31;
  const int wave = tid >> 5;
  const int l16  = lane & 15;
  const int hp   = lane >> 4;           // 16-lane half
  const int mw   = wave & 3;            // 0..3 : 32-row slab
  const int nw   = wave >> 2;           // 0..1 : 32-col slab

  const int m0 = blockIdx.y * 128;
  const int n0 = blockIdx.x * 64;

  const v8f zero = {};
  v8f acc[2][2] = {{zero, zero}, {zero, zero}};

  // async chunk mapping: 16B chunks of 8 halves; A = 512 chunks, B = 256
  const int arow0 = tid >> 2,          ak0 = (tid & 3) * 8;
  const int arow1 = (tid + 256) >> 2,  ak1 = ((tid + 256) & 3) * 8;

  auto issue = [&](int k0, int buf) {
    te_async_b128(&As[buf][tid * 8],
                  &A[(size_t)(m0 + arow0) * K + k0 + ak0]);
    te_async_b128(&As[buf][(tid + 256) * 8],
                  &A[(size_t)(m0 + arow1) * K + k0 + ak1]);
    te_async_b128(&Bst[buf][tid * 8],
                  &Wt[(size_t)(n0 + arow0) * K + k0 + ak0]);
  };

  issue(0, 0);
  int cur = 0;

  for (int k0 = 0; k0 < K; k0 += 32) {
    te_wait_async();       // loads for buf[cur] complete
    __syncthreads();       // everyone done reading buf[cur^1] last iteration
    if (k0 + 32 < K) issue(k0 + 32, cur ^ 1);   // prefetch next slab

    // A fragments: lanes 0-15 row m K=0..15; lanes 16-31 K=16..31 (contiguous)
    v16bf a[2];
#pragma unroll
    for (int tm = 0; tm < 2; ++tm) {
      const bf16* ap = &As[cur][(mw * 32 + tm * 16 + l16) * 32 + hp * 16];
#pragma unroll
      for (int i = 0; i < 8; ++i) { a[tm][i] = ap[i]; a[tm][8 + i] = ap[8 + i]; }
    }
    // B fragments: lane = column n, halves hold K 0..15 / 16..31 (contiguous)
    v16bf bfr[2];
#pragma unroll
    for (int tn = 0; tn < 2; ++tn)
      bfr[tn] = *(const v16bf*)&Bst[cur][(nw * 32 + tn * 16 + l16) * 32 + hp * 16];

#pragma unroll
    for (int tm = 0; tm < 2; ++tm)
#pragma unroll
      for (int tn = 0; tn < 2; ++tn)
        acc[tm][tn] = __builtin_amdgcn_wmma_f32_16x16x32_bf16(
            false, a[tm], false, bfr[tn], (short)0, acc[tm][tn], false, false);

    cur ^= 1;
  }

  // epilogue: C layout — VGPR r: lanes 0-15 row r, lanes 16-31 row r+8
#pragma unroll
  for (int tn = 0; tn < 2; ++tn) {
    const int ncol = n0 + nw * 32 + tn * 16 + l16;
    const float bv = bias ? bias[ncol] : 0.0f;
#pragma unroll
    for (int tm = 0; tm < 2; ++tm) {
#pragma unroll
      for (int r = 0; r < 8; ++r) {
        const int mrow = m0 + mw * 32 + tm * 16 + r + 8 * hp;
        float v = acc[tm][tn][r] + bv;
        if (act == 1)  // exact GELU (erf form)
          v = 0.5f * v * (1.0f + erff(v * 0.70710678118654752f));
        if (Cf) Cf[(size_t)mrow * N + ncol] = v;
        if (Cb) {
          if (cbmode == 0) {
            Cb[(size_t)mrow * N + ncol] = (bf16)v;
          } else {
            // V transposed store: [B,H,DK,S]; consecutive r -> consecutive s
            const int bb = mrow >> 11, ss = mrow & (TE_S - 1);
            const int hh = ncol >> 5,  dd = ncol & (TE_DK - 1);
            Cb[(((size_t)bb * TE_H + hh) * TE_DK + dd) * TE_S + ss] = (bf16)v;
          }
        }
      }
    }
  }
}

// ---------------------------------------------------------------------------
// Flash attention with additive sph bias.
//   Q,K: bf16 [B,S,H,DK] ; Vt: bf16 [B,H,DK,S] ; sph fp32 [B,S,S]
//   Ctx: bf16 [B,S,H,DK].  Block: 128 threads / 4 waves; wave owns 16 q rows.
//   Double-buffered async K/V staging, one barrier per 64-key tile.
// ---------------------------------------------------------------------------
__global__ __launch_bounds__(128) void te_attn(const bf16* __restrict__ Q,
                                               const bf16* __restrict__ Km,
                                               const bf16* __restrict__ Vt,
                                               const float* __restrict__ sph,
                                               bf16* __restrict__ Ctx) {
  __shared__ __align__(32) bf16 Ks[2][64 * 32];    // [key][dk]  2 x 4KB
  __shared__ __align__(32) bf16 Vts[2][32 * 64];   // [dk][key]  2 x 4KB
  __shared__ __align__(32) bf16 Ps[4][16 * 64];    // per-wave P patch 8KB

  const int tid  = threadIdx.x;
  const int lane = tid & 31;
  const int wave = tid >> 5;
  const int l16  = lane & 15;
  const int hp   = lane >> 4;

  const int bz = blockIdx.z;           // batch
  const int hy = blockIdx.y;           // head
  const int q0 = blockIdx.x * 64;      // query tile base
  const float scale = 0.17677669529663687f;   // 1/sqrt(32)

  // Q fragment (A layout) for this wave's 16 rows
  v16bf aq;
  {
    const int qr = q0 + wave * 16 + l16;
    const bf16* qp = Q + (((size_t)bz * TE_S + qr) * TE_H + hy) * TE_DK + hp * 16;
#pragma unroll
    for (int i = 0; i < 8; ++i) { aq[i] = qp[i]; aq[8 + i] = qp[8 + i]; }
  }

  const v8f zero = {};
  v8f o[2] = {zero, zero};
  float mrow[8], lrow[8];
#pragma unroll
  for (int r = 0; r < 8; ++r) { mrow[r] = -1e30f; lrow[r] = 0.0f; }

  auto issueKV = [&](int kb, int buf) {
#pragma unroll
    for (int j = 0; j < 2; ++j) {
      const int c = tid * 2 + j;                 // 0..255 (8-half chunks)
      const int kkey = c >> 2, kdk = (c & 3) * 8;
      te_async_b128(&Ks[buf][c * 8],
                    &Km[(((size_t)bz * TE_S + kb * 64 + kkey) * TE_H + hy) *
                            TE_DK + kdk]);
      const int vrow = c >> 3, vcol = (c & 7) * 8;
      te_async_b128(&Vts[buf][c * 8],
                    &Vt[(((size_t)bz * TE_H + hy) * TE_DK + vrow) * TE_S +
                        kb * 64 + vcol]);
    }
  };

  issueKV(0, 0);
  int cur = 0;

  for (int kb = 0; kb < TE_S / 64; ++kb) {
    te_wait_async();       // K/V for buf[cur] complete
    __syncthreads();       // everyone done reading buf[cur^1] last iteration
    if (kb + 1 < TE_S / 64) issueKV(kb + 1, cur ^ 1);

    // scores S = Q Kt (4 x 16x16 tiles over 64 keys)
    v8f s[4];
#pragma unroll
    for (int nt = 0; nt < 4; ++nt) {
      const v16bf bk = *(const v16bf*)&Ks[cur][(nt * 16 + l16) * 32 + hp * 16];
      v8f acc = zero;
      acc = __builtin_amdgcn_wmma_f32_16x16x32_bf16(
          false, aq, false, bk, (short)0, acc, false, false);
      s[nt] = acc;
    }

    // scale + sph bias, then online softmax per query row
#pragma unroll
    for (int r = 0; r < 8; ++r) {
      const int qg = q0 + wave * 16 + r + 8 * hp;
      const size_t srow = ((size_t)bz * TE_S + qg) * TE_S + kb * 64;
      float mx = -1e30f;
#pragma unroll
      for (int nt = 0; nt < 4; ++nt) {
        float v = s[nt][r] * scale + sph[srow + nt * 16 + l16];
        s[nt][r] = v;
        mx = fmaxf(mx, v);
      }
#pragma unroll
      for (int m = 1; m < 16; m <<= 1) mx = fmaxf(mx, __shfl_xor(mx, m, 32));
      const float mnew  = fmaxf(mrow[r], mx);
      const float alpha = __expf(mrow[r] - mnew);
      float psum = 0.0f;
#pragma unroll
      for (int nt = 0; nt < 4; ++nt) {
        const float p = __expf(s[nt][r] - mnew);
        s[nt][r] = p;
        psum += p;
      }
#pragma unroll
      for (int m = 1; m < 16; m <<= 1) psum += __shfl_xor(psum, m, 32);
      lrow[r] = lrow[r] * alpha + psum;
      mrow[r] = mnew;
      o[0][r] *= alpha;
      o[1][r] *= alpha;
    }

    // re-shape P (C layout) -> A layout through per-wave LDS patch
#pragma unroll
    for (int nt = 0; nt < 4; ++nt)
#pragma unroll
      for (int r = 0; r < 8; ++r)
        Ps[wave][(r + 8 * hp) * 64 + nt * 16 + l16] = (bf16)s[nt][r];

    v16bf a0, a1;
    {
      const bf16* pp = &Ps[wave][l16 * 64 + hp * 16];
#pragma unroll
      for (int i = 0; i < 8; ++i) {
        a0[i] = pp[i];       a0[8 + i] = pp[8 + i];
        a1[i] = pp[32 + i];  a1[8 + i] = pp[40 + i];
      }
    }
    // Ctx += P @ V : B fragments contiguous in Vts ([dk][key])
#pragma unroll
    for (int t = 0; t < 2; ++t) {
      const int dc = t * 16 + l16;
      const v16bf b0 = *(const v16bf*)&Vts[cur][dc * 64 + hp * 16];
      const v16bf b1 = *(const v16bf*)&Vts[cur][dc * 64 + 32 + hp * 16];
      o[t] = __builtin_amdgcn_wmma_f32_16x16x32_bf16(
          false, a0, false, b0, (short)0, o[t], false, false);
      o[t] = __builtin_amdgcn_wmma_f32_16x16x32_bf16(
          false, a1, false, b1, (short)0, o[t], false, false);
    }

    cur ^= 1;
  }

  // normalize and store ctx (bf16, [B,S,H,DK] == reshape(B,S,D))
#pragma unroll
  for (int r = 0; r < 8; ++r) {
    const float inv = 1.0f / lrow[r];
    const int qg = q0 + wave * 16 + r + 8 * hp;
    const size_t base = (((size_t)bz * TE_S + qg) * TE_H + hy) * TE_DK;
#pragma unroll
    for (int t = 0; t < 2; ++t)
      Ctx[base + t * 16 + l16] = (bf16)(o[t][r] * inv);
  }
}

// ---------------------------------------------------------------------------
// Fused residual + LayerNorm:  y = LN(x + s2) * g + b ; also bf16 copy.
// ---------------------------------------------------------------------------
__global__ __launch_bounds__(256) void te_ln(const float* __restrict__ x,
                                             const float* __restrict__ s2,
                                             const float* __restrict__ g,
                                             const float* __restrict__ bt,
                                             float* __restrict__ y,
                                             bf16* __restrict__ yb) {
  __shared__ float red[8];
  const int row = blockIdx.x, t = threadIdx.x;
  const int lane = t & 31, w = t >> 5;
  const size_t idx = (size_t)row * TE_D + t;
  const float v = x[idx] + s2[idx];

  float s = v;
#pragma unroll
  for (int m = 1; m < 32; m <<= 1) s += __shfl_xor(s, m, 32);
  if (lane == 0) red[w] = s;
  __syncthreads();
  float tot = 0.0f;
#pragma unroll
  for (int i = 0; i < 8; ++i) tot += red[i];
  const float mean = tot * (1.0f / TE_D);
  __syncthreads();

  const float d = v - mean;
  s = d * d;
#pragma unroll
  for (int m = 1; m < 32; m <<= 1) s += __shfl_xor(s, m, 32);
  if (lane == 0) red[w] = s;
  __syncthreads();
  tot = 0.0f;
#pragma unroll
  for (int i = 0; i < 8; ++i) tot += red[i];
  const float var = tot * (1.0f / TE_D);

  const float o = d * rsqrtf(var + TE_EPS) * g[t] + bt[t];
  y[idx] = o;
  yb[idx] = (bf16)o;
}

// ---------------------------------------------------------------------------
// Host orchestration
// ---------------------------------------------------------------------------
extern "C" void kernel_launch(void* const* d_in, const int* in_sizes, int n_in,
                              void* d_out, int out_size, void* d_ws,
                              size_t ws_size, hipStream_t stream) {
  (void)in_sizes; (void)n_in; (void)out_size; (void)ws_size;

  const float* src  = (const float*)d_in[0];
  const float* sph  = (const float*)d_in[1];
  const float* Wq   = (const float*)d_in[2];
  const float* bq   = (const float*)d_in[3];
  const float* Wk   = (const float*)d_in[4];
  const float* bk   = (const float*)d_in[5];
  const float* Wv   = (const float*)d_in[6];
  const float* bv   = (const float*)d_in[7];
  const float* Wo   = (const float*)d_in[8];
  const float* bo   = (const float*)d_in[9];
  const float* W1   = (const float*)d_in[10];
  const float* b1   = (const float*)d_in[11];
  const float* W2   = (const float*)d_in[12];
  const float* b2   = (const float*)d_in[13];
  const float* ln1g = (const float*)d_in[14];
  const float* ln1b = (const float*)d_in[15];
  const float* ln2g = (const float*)d_in[16];
  const float* ln2b = (const float*)d_in[17];

  char* w = (char*)d_ws;
  size_t off = 0;
  auto carve = [&](size_t bytes) -> void* {
    void* p = w + off;
    off = (off + bytes + 255) & ~(size_t)255;
    return p;
  };
  const size_t szWp = (size_t)TE_L * TE_D * TE_D;
  const size_t szWf = (size_t)TE_L * TE_D * TE_DFF;
  const size_t szX  = (size_t)TE_M * TE_D;
  const size_t szHh = (size_t)TE_M * TE_DFF;

  bf16* wqb  = (bf16*)carve(szWp * 2);   // transposed [L][N][K]
  bf16* wkb  = (bf16*)carve(szWp * 2);
  bf16* wvb  = (bf16*)carve(szWp * 2);
  bf16* wob  = (bf16*)carve(szWp * 2);
  bf16* w1b  = (bf16*)carve(szWf * 2);
  bf16* w2b  = (bf16*)carve(szWf * 2);
  bf16* xb   = (bf16*)carve(szX * 2);
  bf16* qb   = (bf16*)carve(szX * 2);
  bf16* kbuf = (bf16*)carve(szX * 2);
  bf16* vtb  = (bf16*)carve(szX * 2);    // V transposed [B,H,DK,S]
  bf16* ctxb = (bf16*)carve(szX * 2);
  bf16* hb   = (bf16*)carve(szHh * 2);
  float* s2f = (float*)carve(szX * 4);
  float* xf  = (float*)carve(szX * 4);

  auto cvtT = [&](const float* in, bf16* out, int K, int N) {
    int blocks = (int)(((size_t)TE_L * K * N + 255) / 256);
    te_cvt_t<<<blocks, 256, 0, stream>>>(in, out, K, N);
  };
  cvtT(Wq, wqb, TE_D, TE_D);
  cvtT(Wk, wkb, TE_D, TE_D);
  cvtT(Wv, wvb, TE_D, TE_D);
  cvtT(Wo, wob, TE_D, TE_D);
  cvtT(W1, w1b, TE_D, TE_DFF);
  cvtT(W2, w2b, TE_DFF, TE_D);
  te_cvt<<<(int)((szX + 255) / 256), 256, 0, stream>>>(src, xb, (int)szX);
  hipMemcpyAsync(xf, src, szX * 4, hipMemcpyDeviceToDevice, stream);

  const dim3 gProj(TE_D / 64, TE_M / 128);    // 128-row WG tiles
  const dim3 gFfn1(TE_DFF / 64, TE_M / 128);
  const dim3 gAttn(TE_S / 64, TE_H, TE_B);

  for (int l = 0; l < TE_L; ++l) {
    const bf16* wql = wqb + (size_t)l * TE_D * TE_D;
    const bf16* wkl = wkb + (size_t)l * TE_D * TE_D;
    const bf16* wvl = wvb + (size_t)l * TE_D * TE_D;
    const bf16* wol = wob + (size_t)l * TE_D * TE_D;
    const bf16* w1l = w1b + (size_t)l * TE_D * TE_DFF;   // [DFF][D]
    const bf16* w2l = w2b + (size_t)l * TE_DFF * TE_D;   // [D][DFF]

    te_gemm<<<gProj, 256, 0, stream>>>(xb, wql, bq + (size_t)l * TE_D, nullptr,
                                       qb, TE_M, TE_D, TE_D, 0, 0);
    te_gemm<<<gProj, 256, 0, stream>>>(xb, wkl, bk + (size_t)l * TE_D, nullptr,
                                       kbuf, TE_M, TE_D, TE_D, 0, 0);
    te_gemm<<<gProj, 256, 0, stream>>>(xb, wvl, bv + (size_t)l * TE_D, nullptr,
                                       vtb, TE_M, TE_D, TE_D, 0, 1);
    te_attn<<<gAttn, 128, 0, stream>>>(qb, kbuf, vtb, sph, ctxb);
    te_gemm<<<gProj, 256, 0, stream>>>(ctxb, wol, bo + (size_t)l * TE_D, s2f,
                                       nullptr, TE_M, TE_D, TE_D, 0, 0);
    te_ln<<<TE_M, 256, 0, stream>>>(xf, s2f, ln1g + (size_t)l * TE_D,
                                    ln1b + (size_t)l * TE_D, xf, xb);
    te_gemm<<<gFfn1, 256, 0, stream>>>(xb, w1l, b1 + (size_t)l * TE_DFF,
                                       nullptr, hb, TE_M, TE_DFF, TE_D, 1, 0);
    te_gemm<<<gProj, 256, 0, stream>>>(hb, w2l, b2 + (size_t)l * TE_D, s2f,
                                       nullptr, TE_M, TE_D, TE_DFF, 0, 0);
    float* yout = (l == TE_L - 1) ? (float*)d_out : xf;
    te_ln<<<TE_M, 256, 0, stream>>>(xf, s2f, ln2g + (size_t)l * TE_D,
                                    ln2b + (size_t)l * TE_D, yout, xb);
  }
}
